// GRU_Attention_80977313399718
// MI455X (gfx1250) — matrix operand order repaired and checked by reading, not compiled, not run on hardware
//
#include <hip/hip_runtime.h>
#include <hip/hip_bf16.h>
#include <cstdint>

// ---------------------------------------------------------------------------
// GRU + causal prefix-softmax attention for MI455X (gfx1250, wave32, WMMA).
//
//   K0  f32 -> f16 conversion of x and W_ih (WMMA operands)
//   K1  xw = x @ W_ih^T + b_ih      : WMMA f32_16x16x32_f16, 16x16 tile/wave
//   K2  sequential GRU scan         : ONE workgroup (24 waves, 768 thr),
//                                     W_hh (f16) + h + hw resident in LDS,
//                                     4 WMMAs per wave per step; xw[t+1]
//                                     staged ahead into LDS by the Tensor
//                                     Data Mover (tensor_load_to_lds +
//                                     s_wait_tensorcnt), double-buffered, so
//                                     no global latency on the critical path.
//   K3  e = h @ W_lin + b_lin       : wave-per-row dot + shfl reduce
//   K4  per-batch prefix softmax    : max-reduce, exp, Hillis-Steele cumsum
//   K5  c_out = cumsum(p*h)/Z       : thread per (b,h) running sum
//   K6  alpha_out[t][s] = p_s/Z_t   : bandwidth-bound tril writes (67 MB)
// ---------------------------------------------------------------------------

#define BB 8
#define TT 2048
#define II 64
#define HH 128
#define H3 384
#define MM (BB*TT)                  // 16384 rows
#define TRI ((size_t)TT*(TT+1)/2)   // 2098176

typedef __attribute__((ext_vector_type(16))) _Float16 v16h;
typedef __attribute__((ext_vector_type(8)))  _Float16 v8h;
typedef __attribute__((ext_vector_type(8)))  float    v8f;
typedef unsigned int v4u  __attribute__((ext_vector_type(4)));
typedef int          v8i_ __attribute__((ext_vector_type(8)));
typedef int          v4i_ __attribute__((ext_vector_type(4)));

// gfx1250 transcendental tanh (keeps the gate math off the libm expansion)
__device__ __forceinline__ float fast_tanh(float x) {
    float y;
    asm("v_tanh_f32 %0, %1" : "=v"(y) : "v"(x));
    return y;
}

// ---------------------------------------------------------------- K0: f32->f16
__global__ void k_cvt_f16(const float* __restrict__ src,
                          _Float16* __restrict__ dst, int n) {
    int i = blockIdx.x * blockDim.x + threadIdx.x;
    if (i < n) dst[i] = (_Float16)src[i];
}

// ------------------------------------------------- K1: xw = x @ W_ih^T + b_ih
// grid (M/16, H3/16), block 32 (one wave per 16x16 output tile).
// A frag (16x32 f16): lane<16: halves 0..7 = K+0..7, 8..15 = K+16..23
//                     lane>=16: halves 0..7 = K+8..15, 8..15 = K+24..31
// B frag (32x16 f16): N = lane%16; same K order as the A frag:
//                     lane<16: K+0..7, K+16..23; lane>=16: K+8..15, K+24..31.  B[k][n] = W_ih[n][k] so a
//                     B frag is two 8-half runs (16 apart) of W_ih row n.
__global__ __launch_bounds__(32)
void k_proj(const _Float16* __restrict__ xh,    // (M, 64)
            const _Float16* __restrict__ wih,   // (384, 64)
            const float*    __restrict__ bih,   // (384)
            float*          __restrict__ xw) {  // (M, 384)
    const int m0 = blockIdx.x * 16;
    const int n0 = blockIdx.y * 16;
    const int lane = threadIdx.x & 31;
    const int r    = lane & 15;
    const int aoff = (lane < 16) ? 0 : 8;
    const int boff = (lane < 16) ? 0 : 8;
    const _Float16* aRow = xh  + (size_t)(m0 + r) * II;
    const _Float16* bRow = wih + (size_t)(n0 + r) * II;

    v8f acc = {};
#pragma unroll
    for (int ks = 0; ks < 2; ++ks) {            // K = 64 -> 2 steps of 32
        const int kb = ks * 32;
        v8h alo = *(const v8h*)(aRow + kb + aoff);
        v8h ahi = *(const v8h*)(aRow + kb + aoff + 16);
        v8h blo = *(const v8h*)(bRow + kb + boff);
        v8h bhi = *(const v8h*)(bRow + kb + boff + 16);
        v16h a, b;
#pragma unroll
        for (int i = 0; i < 8; ++i) {
            a[i] = alo[i]; a[8+i] = ahi[i];
            b[i] = blo[i]; b[8+i] = bhi[i];
        }
        acc = __builtin_amdgcn_wmma_f32_16x16x32_f16(
            false, a, false, b, (short)0, acc, false, false);
    }
    const float bias = bih[n0 + r];
#pragma unroll
    for (int i = 0; i < 8; ++i) {               // C: VGPR i -> M=i / i+8
        const int m = m0 + i + ((lane < 16) ? 0 : 8);
        xw[(size_t)m * H3 + n0 + r] = acc[i] + bias;
    }
}

// -------------------------------------------------------------- K2: GRU scan
// One workgroup, 24 waves. Wave j owns output columns [16j,16j+16) of the
// 8x384 hw = h @ W_hh^T.  A rows = 8 batches (rows 8..15 stay zero).
//
// LDS map (dynamic, from offset 0):
//   [0, 24576)        sxw double buffer  2 x (8 x 384) f32  <- TDM target
//   [24576, ...)      sW f16 [384][136], sh f16 [16][136], shw f32 [8][384],
//                     sh32 f32 [8][128], sb f32 [384]
#define WSTRIDE   136                     // f16 row stride (16B aligned)
#define SXW_BYTES (2 * 8 * H3 * 4)        // 24576

// Issue one TDM descriptor: copy the 2D tile { 8 rows x 384 f32,
// row stride T*384 elements } starting at gsrc into LDS byte offset lds_off.
__device__ __forceinline__ void tdm_load_xw(const float* gsrc,
                                            unsigned lds_off) {
    const unsigned long long ga = (unsigned long long)(uintptr_t)gsrc;
    const v4u g0 = { 1u,                                  // count = 1 (valid)
                     lds_off,                             // lds_addr (bytes)
                     (unsigned)(ga & 0xffffffffu),        // global_addr lo
                     (unsigned)((ga >> 32) & 0x01ffffffu) // global_addr hi
                         | (2u << 30) };                  // type = 2 (image)
    const v8i_ g1 = { (int)(2u << 16),      // data_size = 2 (4 bytes)
                      (int)((unsigned)H3 << 16),  // tensor_dim0[15:0] @ 63:48
                      (int)(8u << 16),      // dim0 hi | tensor_dim1[15:0]
                      (int)((unsigned)H3 << 16),  // dim1 hi | tile_dim0
                      8,                    // tile_dim1 = 8, tile_dim2 = 0
                      TT * H3,              // tensor_dim0_stride lo32
                      0, 0 };               // stride hi / dim1_stride
    const v4i_ gz = { 0, 0, 0, 0 };
#if __clang_major__ >= 23
    const v8i_ gz8 = { 0, 0, 0, 0, 0, 0, 0, 0 };
    __builtin_amdgcn_tensor_load_to_lds(g0, g1, gz, gz, gz8, 0);
#else
    __builtin_amdgcn_tensor_load_to_lds(g0, g1, gz, gz, 0);
#endif
}

__global__ __launch_bounds__(768, 1)
void k_gru_scan(const float* __restrict__ xw,    // (M, 384) incl. b_ih
                const float* __restrict__ whh,   // (384, 128)
                const float* __restrict__ bhh,   // (384)
                float*       __restrict__ hout)  // (B, T, 128)
{
    extern __shared__ char smem[];
    float*    sxw  = (float*)smem;                    // [2][8][384] TDM stage
    _Float16* sW   = (_Float16*)(smem + SXW_BYTES);   // [384][136] f16
    _Float16* sh   = sW + 384 * WSTRIDE;              // [16][136]  f16 (A)
    float*    shw  = (float*)(sh + 16 * WSTRIDE);     // [8][384]
    float*    sh32 = shw + 8 * H3;                    // [8][128]  h in f32
    float*    sb   = sh32 + 8 * HH;                   // [384]

    const int tid = threadIdx.x;
    for (int i = tid; i < H3 * HH; i += 768) {
        sW[(i >> 7) * WSTRIDE + (i & 127)] = (_Float16)whh[i];
    }
    for (int i = tid; i < 16 * WSTRIDE; i += 768) sh[i] = (_Float16)0.f;
    for (int i = tid; i < 8 * HH;       i += 768) sh32[i] = 0.f;
    for (int i = tid; i < H3;           i += 768) sb[i] = bhh[i];

    const int lane = tid & 31;
    const int wv   = tid >> 5;
    const int n0   = wv * 16;                         // wave's N-tile
    const int r    = lane & 15;
    const int aoff = (lane < 16) ? 0 : 8;
    const int boff = (lane < 16) ? 0 : 8;
    const _Float16* aRow = sh + r * WSTRIDE;
    const _Float16* bRow = sW + (n0 + r) * WSTRIDE;

    if (wv == 0) tdm_load_xw(xw, 0u);                 // stage xw[t=0] -> buf 0
    __syncthreads();

    for (int t = 0; t < TT; ++t) {
        if (wv == 0) {
            // stage xw[t+1] into the other buffer, then retire load t.
            // TDM ops of one wave complete in order, so tensorcnt<=1 means
            // the older (step-t) descriptor has fully landed in LDS.
            if (t + 1 < TT) {
                tdm_load_xw(xw + (size_t)(t + 1) * H3,
                            (unsigned)(((t + 1) & 1) * (8 * H3 * 4)));
                __builtin_amdgcn_s_wait_tensorcnt((short)1);
            } else {
                __builtin_amdgcn_s_wait_tensorcnt((short)0);
            }
        }
        v8f acc = {};
#pragma unroll
        for (int ks = 0; ks < 4; ++ks) {              // K = 128
            const int kb = ks * 32;
            v8h alo = *(const v8h*)(aRow + kb + aoff);
            v8h ahi = *(const v8h*)(aRow + kb + aoff + 16);
            v8h blo = *(const v8h*)(bRow + kb + boff);
            v8h bhi = *(const v8h*)(bRow + kb + boff + 16);
            v16h a, b;
#pragma unroll
            for (int i = 0; i < 8; ++i) {
                a[i] = alo[i]; a[8+i] = ahi[i];
                b[i] = blo[i]; b[8+i] = bhi[i];
            }
            acc = __builtin_amdgcn_wmma_f32_16x16x32_f16(
                false, a, false, b, (short)0, acc, false, false);
        }
        if (lane < 16) {                              // M rows 0..7 are real
#pragma unroll
            for (int i = 0; i < 8; ++i) shw[i * H3 + n0 + lane] = acc[i];
        }
        __syncthreads();   // publishes hw AND the TDM-staged xw[t] tile

        // gate math: 1024 (b,h) elements over 768 threads, all LDS-resident
        const float* xwt0 = sxw + (t & 1) * (8 * H3);
        for (int e = tid; e < BB * HH; e += 768) {
            const int b = e >> 7, hh = e & 127;
            const float* xwt = xwt0 + b * H3;
            const float xr = xwt[hh], xz = xwt[HH + hh], xn = xwt[2*HH + hh];
            const float hr = shw[b*H3 +        hh] + sb[hh];
            const float hz = shw[b*H3 + HH  +  hh] + sb[HH + hh];
            const float hn = shw[b*H3 + 2*HH + hh] + sb[2*HH + hh];
            const float rg = 1.f / (1.f + __expf(-(xr + hr)));
            const float zg = 1.f / (1.f + __expf(-(xz + hz)));
            const float ng = fast_tanh(xn + rg * hn);
            const float hp = sh32[b * HH + hh];
            const float hv = (1.f - zg) * ng + zg * hp;
            sh32[b * HH + hh] = hv;
            sh[b * WSTRIDE + hh] = (_Float16)hv;      // A for next step
            hout[((size_t)b * TT + t) * HH + hh] = hv;
        }
        __syncthreads();
    }
}

// ------------------------------------------------- K3: e = h @ W_lin + b_lin
__global__ __launch_bounds__(256)
void k_logits(const float* __restrict__ h, const float* __restrict__ wlin,
              const float* __restrict__ blin, float* __restrict__ e) {
    const int lane = threadIdx.x & 31;
    const int row  = blockIdx.x * 8 + (threadIdx.x >> 5);   // one wave/row
    float s = 0.f;
#pragma unroll
    for (int k = lane; k < HH; k += 32) s += h[(size_t)row * HH + k] * wlin[k];
#pragma unroll
    for (int m = 16; m >= 1; m >>= 1) s += __shfl_xor(s, m, 32);
    if (lane == 0) e[row] = s + blin[0];
}

// --------------------------------------- K4: per-batch prefix softmax tables
__global__ __launch_bounds__(256)
void k_prefix_softmax(const float* __restrict__ e,
                      float* __restrict__ p, float* __restrict__ invZ) {
    __shared__ float se[TT];
    __shared__ float red[256];
    const int b = blockIdx.x, tid = threadIdx.x;

    float lmax = -INFINITY;
    for (int t = tid; t < TT; t += 256) {
        const float v = e[b * TT + t];
        se[t] = v; lmax = fmaxf(lmax, v);
    }
    red[tid] = lmax; __syncthreads();
    for (int s = 128; s > 0; s >>= 1) {
        if (tid < s) red[tid] = fmaxf(red[tid], red[tid + s]);
        __syncthreads();
    }
    const float m = red[0];
    __syncthreads();

    float vals[8]; float run = 0.f;                  // 8 consecutive per thread
#pragma unroll
    for (int i = 0; i < 8; ++i) {
        const int t = tid * 8 + i;
        const float pv = __expf(se[t] - m);
        vals[i] = pv; run += pv; p[b * TT + t] = pv;
    }
    red[tid] = run; __syncthreads();
    for (int off = 1; off < 256; off <<= 1) {        // Hillis-Steele inclusive
        const float v   = red[tid];
        const float add = (tid >= off) ? red[tid - off] : 0.f;
        __syncthreads();
        red[tid] = v + add;
        __syncthreads();
    }
    float c = (tid == 0) ? 0.f : red[tid - 1];
#pragma unroll
    for (int i = 0; i < 8; ++i) {
        c += vals[i];
        invZ[b * TT + tid * 8 + i] = 1.f / c;
    }
}

// ---------------------------------------------- K5: c_out = cumsum(p*h) / Z
__global__ __launch_bounds__(256)
void k_context(const float* __restrict__ h, const float* __restrict__ p,
               const float* __restrict__ invZ, float* __restrict__ cout) {
    const int g  = blockIdx.x * 256 + threadIdx.x;   // 1024 threads total
    const int b  = g >> 7, hh = g & 127;
    float s = 0.f;
    for (int t = 0; t < TT; ++t) {
        const size_t idx = ((size_t)b * TT + t) * HH + hh;
        s += p[b * TT + t] * h[idx];
        cout[idx] = s * invZ[b * TT + t];
    }
}

// ------------------------------------------------- K6: tril alpha_out writes
__global__ __launch_bounds__(256)
void k_alpha(const float* __restrict__ p, const float* __restrict__ invZ,
             float* __restrict__ alpha) {
    const int t = blockIdx.x, b = blockIdx.y;
    const float iz = invZ[b * TT + t];
    float* dst = alpha + (size_t)b * TRI + (size_t)t * (t + 1) / 2;
    for (int s = threadIdx.x; s <= t; s += 256) dst[s] = p[b * TT + s] * iz;
}

// ---------------------------------------------------------------------------
extern "C" void kernel_launch(void* const* d_in, const int* in_sizes, int n_in,
                              void* d_out, int out_size, void* d_ws, size_t ws_size,
                              hipStream_t stream) {
    const float* x    = (const float*)d_in[0];   // (8, 2048, 64)
    const float* Wih  = (const float*)d_in[1];   // (384, 64)
    const float* Whh  = (const float*)d_in[2];   // (384, 128)
    const float* bih  = (const float*)d_in[3];   // (384)
    const float* bhh  = (const float*)d_in[4];   // (384)
    const float* Wlin = (const float*)d_in[5];   // (1, 128)
    const float* blin = (const float*)d_in[6];   // (1)

    float* cout  = (float*)d_out;                       // (8, 2048, 128)
    float* alpha = (float*)d_out + (size_t)MM * HH;     // (8, TRI, 1)

    // workspace carve
    char* w = (char*)d_ws;
    _Float16* xh   = (_Float16*)w;  w += (size_t)MM * II * 2;   // 2 MB
    _Float16* wihh = (_Float16*)w;  w += (size_t)H3 * II * 2;   // 48 KB
    float* xw   = (float*)w;        w += (size_t)MM * H3 * 4;   // 24 MB
    float* hbuf = (float*)w;        w += (size_t)MM * HH * 4;   // 8 MB
    float* e    = (float*)w;        w += (size_t)MM * 4;
    float* p    = (float*)w;        w += (size_t)MM * 4;
    float* invZ = (float*)w;        w += (size_t)MM * 4;
    (void)ws_size; (void)in_sizes; (void)n_in; (void)out_size;

    // K0: conversions
    k_cvt_f16<<<(MM * II + 255) / 256, 256, 0, stream>>>(x, xh, MM * II);
    k_cvt_f16<<<(H3 * II + 255) / 256, 256, 0, stream>>>(Wih, wihh, H3 * II);

    // K1: input projection (WMMA)
    k_proj<<<dim3(MM / 16, H3 / 16), 32, 0, stream>>>(xh, wihh, bih, xw);

    // K2: sequential GRU scan (single persistent workgroup, WMMA + TDM + LDS)
    const size_t lds = (size_t)SXW_BYTES               // TDM double buffer
                     + (size_t)384 * WSTRIDE * 2       // sW
                     + (size_t)16  * WSTRIDE * 2       // sh
                     + (size_t)8 * H3 * 4              // shw
                     + (size_t)8 * HH * 4              // sh32
                     + (size_t)H3 * 4;                 // sb  (~148 KB of 320)
    k_gru_scan<<<1, 768, lds, stream>>>(xw, Whh, bhh, hbuf);

    // K3..K6: attention via prefix softmax
    k_logits<<<MM / 8, 256, 0, stream>>>(hbuf, Wlin, blin, e);
    k_prefix_softmax<<<BB, 256, 0, stream>>>(e, p, invZ);
    k_context<<<(BB * HH) / 256, 256, 0, stream>>>(hbuf, p, invZ, cout);
    k_alpha<<<dim3(TT, BB), 256, 0, stream>>>(p, invZ, alpha);
}
